// GraphEdgeActionGNN_44659069944306
// MI455X (gfx1250) — matrix operand order repaired and checked by reading, compiled, not run on hardware
//
#include <hip/hip_runtime.h>

#define NN 128   // nodes per batch
#define DD 128   // feature dim
#define NB 2048  // batches
#define EPB 1024 // edges per batch

typedef _Float16 v16h __attribute__((ext_vector_type(16)));
typedef _Float16 v8h  __attribute__((ext_vector_type(8)));
typedef float    v8f  __attribute__((ext_vector_type(8)));

// A fragment: 16x32 f16 tile, rows m = lane%16.
// lanes 0-15 hold K = kb+{0..7, 16..23}; lanes 16-31 hold K = kb+{8..15, 24..31}
__device__ __forceinline__ v16h frag_a(const _Float16* base, int row, int kb, int hi) {
    const v8h* p = (const v8h*)(base + row * DD + kb + hi * 8);
    v8h x = p[0];
    v8h y = p[2]; // +16 elements
    v16h r;
#pragma unroll
    for (int i = 0; i < 8; ++i) { r[i] = x[i]; r[i + 8] = y[i]; }
    return r;
}

// B fragment: 32x16 f16 tile, cols n = lane%16 (source stored n-major, k contiguous).
// lanes 0-15 hold K = kb+{0..7, 16..23}; lanes 16-31 hold K = kb+{8..15, 24..31}
__device__ __forceinline__ v16h frag_b(const _Float16* base, int row, int kb, int hi) {
    const v8h* p = (const v8h*)(base + row * DD + kb + hi * 8);
    v8h x = p[0];
    v8h y = p[2];
    v16h r;
#pragma unroll
    for (int i = 0; i < 8; ++i) { r[i] = x[i]; r[i + 8] = y[i]; }
    return r;
}

// One wave computes a 16x128 strip: 8 output tiles, K-loop of 4 x 32.
__device__ __forceinline__ void gemm128(const _Float16* A, const _Float16* Bm,
                                        int m0, int lo, int hi, v8f acc[8]) {
#pragma unroll
    for (int kb = 0; kb < DD; kb += 32) {
        v16h a = frag_a(A, m0 + lo, kb, hi);
#pragma unroll
        for (int t = 0; t < 8; ++t) {
            v16h b = frag_b(Bm, t * 16 + lo, kb, hi);
            acc[t] = __builtin_amdgcn_wmma_f32_16x16x32_f16(
                false, a, false, b, (short)0, acc[t], false, false);
        }
    }
}

// Sum across the 16-lane group (bits 0-3 of lane id); wave32.
__device__ __forceinline__ float rowred16(float v) {
#pragma unroll
    for (int m = 1; m < 16; m <<= 1) v += __shfl_xor(v, m, 32);
    return v;
}

// bias (+ optional LayerNorm + optional ReLU), convert to f16, store to Xa.
// C/D layout: lane holds rows {8*hi + 0..7} x cols {16t + lane%16}.
__device__ __forceinline__ void epilogue(v8f acc[8], const float* bias,
                                         const float* lng, const float* lnb,
                                         bool doLN, bool doRelu,
                                         _Float16* Xa, int m0, int lo, int hi) {
    float bs[8], gg[8], bb[8];
#pragma unroll
    for (int t = 0; t < 8; ++t) {
        int col = t * 16 + lo;
        bs[t] = bias[col];
        gg[t] = doLN ? lng[col] : 1.f;
        bb[t] = doLN ? lnb[col] : 0.f;
    }
#pragma unroll
    for (int r = 0; r < 8; ++r) {
        float x[8];
        float s = 0.f;
#pragma unroll
        for (int t = 0; t < 8; ++t) { x[t] = acc[t][r] + bs[t]; s += x[t]; }
        if (doLN) {
            float mu = rowred16(s) * (1.f / 128.f);
            float d = 0.f;
#pragma unroll
            for (int t = 0; t < 8; ++t) { float u = x[t] - mu; d += u * u; }
            float var = rowred16(d) * (1.f / 128.f);
            float rs = rsqrtf(var + 1e-5f);
#pragma unroll
            for (int t = 0; t < 8; ++t) x[t] = (x[t] - mu) * rs * gg[t] + bb[t];
        }
        if (doRelu) {
#pragma unroll
            for (int t = 0; t < 8; ++t) x[t] = fmaxf(x[t], 0.f);
        }
        int row = m0 + r + hi * 8;
#pragma unroll
        for (int t = 0; t < 8; ++t) Xa[row * DD + t * 16 + lo] = (_Float16)x[t];
    }
}

// Stage W^T (n-major, f16) into LDS: Wt[n*128+k] = W[k*128+n]. LDS writes coalesced.
__device__ __forceinline__ void load_wT(_Float16* Wt, const float* W, int tid) {
    for (int i = tid; i < DD * DD; i += 256)
        Wt[i] = (_Float16)W[(i & 127) * DD + (i >> 7)];
}

__global__ void __launch_bounds__(256)
gnn_fused(const int* __restrict__ nf, const int* __restrict__ src,
          const int* __restrict__ dst,
          const float* embed, const float* gw1, const float* gb1,
          const float* glng, const float* glnb, const float* gw2, const float* gb2,
          const float* sw1, const float* sb1, const float* sw2, const float* sb2,
          const float* ng, const float* nbv,
          const float* ew1, const float* eb1, const float* elng, const float* elnb,
          const float* ew2, const float* eb2,
          float* __restrict__ out) {
    __shared__ __align__(128) char pool[65536];
    _Float16* Xa  = (_Float16*)pool;            // 32KB activations (rows wave-private)
    unsigned* Cb  = (unsigned*)pool;            // aliased: edge-count histogram
    _Float16* Wt  = (_Float16*)(pool + 32768);  // 32KB weight^T staging
    float*    scr = (float*)(pool + 32768);     // reused for exit head

    const int b    = blockIdx.x;
    const int tid  = threadIdx.x;
    const int lane = tid & 31;
    const int lo   = lane & 15;
    const int hi   = lane >> 4;
    const int m0   = (tid >> 5) * 16;

    // ---- Stage 0: zero counts, stage embed^T
    for (int i = tid; i < DD * DD / 2; i += 256) Cb[i] = 0u;
    load_wT(Wt, embed, tid);
    __syncthreads();

    // ---- Edge histogram: C[dst_local][feat(src)]++, plus self term (h = agg + x).
    // Two u16 counters per u32 cell; counts <= 1025 so no carry into high half.
    for (int e = tid; e < EPB; e += 256) {
        int sg = src[b * EPB + e];
        int dl = dst[b * EPB + e] - b * NN;
        int sf = nf[sg];
        atomicAdd(&Cb[dl * 64 + (sf >> 1)], 1u << (16 * (sf & 1)));
    }
    if (tid < NN) {
        int sf = nf[b * NN + tid];
        atomicAdd(&Cb[tid * 64 + (sf >> 1)], 1u << (16 * (sf & 1)));
    }
    __syncthreads();

    // ---- Convert u16 counts -> f16 in place (exact: counts < 2048)
    for (int i = tid; i < DD * DD / 2; i += 256) {
        unsigned c = Cb[i];
        union { _Float16 h[2]; unsigned u; } pk;
        pk.h[0] = (_Float16)(float)(c & 0xffffu);
        pk.h[1] = (_Float16)(float)(c >> 16);
        Cb[i] = pk.u;
    }
    __syncthreads();

    v8f acc[8];
    const v8f vzero = {};

    // ---- GEMM 1: h0 = C @ embed. A aliases Xa: each wave's A reads (its own 16
    // rows) complete, in DS order, before it stores h0 into those same rows.
#pragma unroll
    for (int t = 0; t < 8; ++t) acc[t] = vzero;
    gemm128((const _Float16*)Cb, Wt, m0, lo, hi, acc);
#pragma unroll
    for (int r = 0; r < 8; ++r) {
        int row = m0 + r + hi * 8;
#pragma unroll
        for (int t = 0; t < 8; ++t) Xa[row * DD + t * 16 + lo] = (_Float16)acc[t][r];
    }
    __syncthreads();
    load_wT(Wt, gw1, tid);
    __syncthreads();

    // ---- GEMM 2: relu(LN(h0 @ gin_w1 + gin_b1))
#pragma unroll
    for (int t = 0; t < 8; ++t) acc[t] = vzero;
    gemm128(Xa, Wt, m0, lo, hi, acc);
    epilogue(acc, gb1, glng, glnb, true, true, Xa, m0, lo, hi);
    __syncthreads();
    load_wT(Wt, gw2, tid);
    __syncthreads();

    // ---- GEMM 3: @ gin_w2 + gin_b2
#pragma unroll
    for (int t = 0; t < 8; ++t) acc[t] = vzero;
    gemm128(Xa, Wt, m0, lo, hi, acc);
    epilogue(acc, gb2, glng, glnb, false, false, Xa, m0, lo, hi);
    __syncthreads();
    load_wT(Wt, sw1, tid);
    __syncthreads();

    // ---- GEMM 4: relu(@ seq_w1 + seq_b1)
#pragma unroll
    for (int t = 0; t < 8; ++t) acc[t] = vzero;
    gemm128(Xa, Wt, m0, lo, hi, acc);
    epilogue(acc, sb1, glng, glnb, false, true, Xa, m0, lo, hi);
    __syncthreads();
    load_wT(Wt, sw2, tid);
    __syncthreads();

    // ---- GEMM 5: Xn = LN(@ seq_w2 + seq_b2, norm_g/b)
#pragma unroll
    for (int t = 0; t < 8; ++t) acc[t] = vzero;
    gemm128(Xa, Wt, m0, lo, hi, acc);
    epilogue(acc, sb2, ng, nbv, true, false, Xa, m0, lo, hi);
    __syncthreads(); // Xn visible to all waves; Wt region now free -> scr

    // ---- Exit head: means -> LN(means@exit_w1+b1) -> relu -> @exit_w2 + b2
    if (tid < DD) {
        float s = 0.f;
        for (int r = 0; r < NN; ++r) s += (float)Xa[r * DD + tid];
        scr[tid] = s * (1.f / 128.f);
    }
    __syncthreads();
    if (tid < DD) {
        float t = eb1[tid];
        for (int k = 0; k < DD; ++k) t = fmaf(scr[k], ew1[k * DD + tid], t);
        scr[DD + tid] = t;
    }
    __syncthreads();
    if (tid < DD) {
        float mu = 0.f;
        for (int k = 0; k < DD; ++k) mu += scr[DD + k];
        mu *= (1.f / 128.f);
        float var = 0.f;
        for (int k = 0; k < DD; ++k) { float u = scr[DD + k] - mu; var += u * u; }
        var *= (1.f / 128.f);
        float e = (scr[DD + tid] - mu) * rsqrtf(var + 1e-5f) * elng[tid] + elnb[tid];
        e = fmaxf(e, 0.f);
        scr[2 * DD + tid] = e * ew2[tid];
    }
    __syncthreads();
    if (tid == 0) {
        float ea = eb2[0];
        for (int k = 0; k < DD; ++k) ea += scr[2 * DD + k];
        out[(size_t)b * 3 + 0] = -__builtin_inff();
        out[(size_t)b * 3 + 1] = 0.f;
        out[(size_t)b * 3 + 2] = ea;
        out[3 * NB + b] = 0.f; // edge_class
        out[4 * NB + b] = 0.f; // node_class
    }

    // ---- GEMM 6: Gram = Xn @ Xn^T / sqrt(D); B fragments read Xn row-major
    // directly (B[k][n] = Xn[n][k] means the n-major source IS Xn).
#pragma unroll
    for (int t = 0; t < 8; ++t) acc[t] = vzero;
    gemm128(Xa, Xa, m0, lo, hi, acc);
    const float inv_sqrt_d = 0.08838834764831845f;
    float* ea_out = out + 5 * NB + (size_t)b * 8128;
#pragma unroll
    for (int r = 0; r < 8; ++r) {
        int i = m0 + r + hi * 8;
#pragma unroll
        for (int t = 0; t < 8; ++t) {
            int j = t * 16 + lo;
            if (j > i) {
                int idx = i * 127 - (i * (i - 1)) / 2 + (j - i - 1);
                ea_out[idx] = acc[t][r] * inv_sqrt_d;
            }
        }
    }
}

extern "C" void kernel_launch(void* const* d_in, const int* in_sizes, int n_in,
                              void* d_out, int out_size, void* d_ws, size_t ws_size,
                              hipStream_t stream) {
    (void)in_sizes; (void)n_in; (void)d_ws; (void)ws_size; (void)out_size;
    const int* nf  = (const int*)d_in[0];
    const int* ei  = (const int*)d_in[1];
    const int* src = ei;
    const int* dst = ei + (size_t)NB * EPB;

    gnn_fused<<<NB, 256, 0, stream>>>(
        nf, src, dst,
        (const float*)d_in[2],                       // embed
        (const float*)d_in[3],  (const float*)d_in[4],   // gin_w1, gin_b1
        (const float*)d_in[5],  (const float*)d_in[6],   // gin_ln_g, gin_ln_b
        (const float*)d_in[7],  (const float*)d_in[8],   // gin_w2, gin_b2
        (const float*)d_in[9],  (const float*)d_in[10],  // seq_w1, seq_b1
        (const float*)d_in[11], (const float*)d_in[12],  // seq_w2, seq_b2
        (const float*)d_in[13], (const float*)d_in[14],  // norm_g, norm_b
        (const float*)d_in[15], (const float*)d_in[16],  // exit_w1, exit_b1
        (const float*)d_in[17], (const float*)d_in[18],  // exit_ln_g, exit_ln_b
        (const float*)d_in[19], (const float*)d_in[20],  // exit_w2, exit_b2
        (float*)d_out);
}
